// Attention_36910948942459
// MI455X (gfx1250) — matrix operand order repaired and checked by reading, not compiled, not run on hardware
//
#include <hip/hip_runtime.h>
#include <hip/hip_bf16.h>
#include <cstdint>

#define B_   2
#define S_   2048
#define D_   2048
#define H_   16
#define HK_  4
#define DK_  128
#define REP_ 4
#define KD_  (HK_ * DK_)   // 512

// LDS row paddings (bank-conflict avoidance; 64 banks x 4B)
#define PAD32  40    // 32-elem rows -> 80B stride (20 banks)
#define PAD128 136   // 128-elem rows -> 272B stride (4 banks offset per row)

typedef __bf16 bf16_t;
typedef __attribute__((ext_vector_type(16))) __bf16 v16bf;
typedef __attribute__((ext_vector_type(8)))  __bf16 v8bf;
typedef __attribute__((ext_vector_type(8)))  float  v8f;

__device__ __forceinline__ v8f vzero8() {
  v8f v;
#pragma unroll
  for (int i = 0; i < 8; ++i) v[i] = 0.0f;
  return v;
}

__device__ __forceinline__ v8f wmma_bf16(v16bf a, v16bf b, v8f c) {
  return __builtin_amdgcn_wmma_f32_16x16x32_bf16(false, a, false, b, (short)0, c,
                                                 false, false);
}

// ---------------- CDNA5 async copy: global -> LDS (ASYNCcnt tracked) ---------
__device__ __forceinline__ unsigned lds_addr_of(const void* p) {
  return (unsigned)(unsigned long long)(__attribute__((address_space(3))) const char*)p;
}

__device__ __forceinline__ void async_copy_b128(const void* gsrc, void* ldst) {
  asm volatile("global_load_async_to_lds_b128 %0, %1, off"
               :
               : "v"(lds_addr_of(ldst)), "v"((unsigned long long)(size_t)gsrc)
               : "memory");
}

#define ASYNC_WAIT0() asm volatile("s_wait_asynccnt 0x0" ::: "memory")
#define DS_WAIT0()    asm volatile("s_wait_dscnt 0x0" ::: "memory")

// ---------------- fragment loads (generic pointer: LDS or global) -----------
// A fragment: 16x32 (MxK); lanes 0-15: M=lane, K 0..7 & 16..23; lanes 16-31: K 8..15 & 24..31.
__device__ __forceinline__ v16bf frag_a_ld(const bf16_t* base, int ld, int lane) {
  int row  = lane & 15;
  int koff = (lane >> 4) << 3;                 // 0 or 8
  const bf16_t* p = base + (size_t)row * ld + koff;
  v8bf lo = *(const v8bf*)p;
  v8bf hi = *(const v8bf*)(p + 16);
  v16bf f;
#pragma unroll
  for (int i = 0; i < 8; ++i) { f[i] = lo[i]; f[i + 8] = hi[i]; }
  return f;
}

// B fragment: 32x16 (KxN); column n stored contiguous as memory row n (stride ld).
// lanes 0-15: N=lane, K 0..7 & 16..23; lanes 16-31: K 8..15 & 24..31.
__device__ __forceinline__ v16bf frag_b_ld(const bf16_t* base, int ld, int lane) {
  int col  = lane & 15;
  int koff = (lane >> 4) << 3;                 // 0 or 8
  const bf16_t* p = base + (size_t)col * ld + koff;
  v8bf lo = *(const v8bf*)p;
  v8bf hi = *(const v8bf*)(p + 16);
  v16bf f;
#pragma unroll
  for (int i = 0; i < 8; ++i) { f[i] = lo[i]; f[i + 8] = hi[i]; }
  return f;
}

// ---------------------------------------------------------------------------
__global__ void cvt_f32_bf16(const float* __restrict__ in, bf16_t* __restrict__ out,
                             size_t n) {
  size_t i = (size_t)blockIdx.x * blockDim.x + threadIdx.x;
  if (i < n) out[i] = (bf16_t)in[i];
}

// ---------------------------------------------------------------------------
// C[M,N] = A[M,K] @ W[N,K]^T + bias; async-to-LDS double buffering.
// Block: 256 threads = 8 waves; block tile 128x128; wave tile 16x128; K-step 32.
__global__ void __launch_bounds__(256)
gemm_bf16_nt(const bf16_t* __restrict__ A, const bf16_t* __restrict__ W,
             const float* __restrict__ bias, float* __restrict__ Cf,
             bf16_t* __restrict__ Cb, int M, int N, int K) {
  __shared__ bf16_t As[2][128][PAD32];
  __shared__ bf16_t Bs[2][128][PAD32];

  int tid  = threadIdx.x;
  int lane = tid & 31;
  int wave = tid >> 5;
  int n0 = blockIdx.x * 128;
  int m0 = blockIdx.y * 128;

  // stage a 128x32 tile (row stride ldg): 512 16B chunks, 2 per thread
  auto stage = [&](bf16_t(*dst)[PAD32], const bf16_t* src, int ldg) {
#pragma unroll
    for (int j = 0; j < 2; ++j) {
      int idx = tid + j * 256;
      int row = idx >> 2;          // 4 chunks of 8 bf16 per row
      int ch  = idx & 3;
      async_copy_b128(src + (size_t)row * ldg + ch * 8, &dst[row][ch * 8]);
    }
  };

  v8f acc[8];
#pragma unroll
  for (int t = 0; t < 8; ++t) acc[t] = vzero8();

  stage(As[0], A + (size_t)m0 * K, K);
  stage(Bs[0], W + (size_t)n0 * K, K);

  int nks = K / 32;
  for (int ks = 0; ks < nks; ++ks) {
    int cur = ks & 1;
    ASYNC_WAIT0();
    __syncthreads();
    if (ks + 1 < nks) {
      stage(As[cur ^ 1], A + (size_t)m0 * K + (size_t)(ks + 1) * 32, K);
      stage(Bs[cur ^ 1], W + (size_t)n0 * K + (size_t)(ks + 1) * 32, K);
    }
    // batch all fragment loads, then issue the 8 WMMAs back-to-back
    v16bf a = frag_a_ld(&As[cur][wave * 16][0], PAD32, lane);
    v16bf bfr[8];
#pragma unroll
    for (int t = 0; t < 8; ++t) bfr[t] = frag_b_ld(&Bs[cur][t * 16][0], PAD32, lane);
#pragma unroll
    for (int t = 0; t < 8; ++t) acc[t] = wmma_bf16(a, bfr[t], acc[t]);
  }

  int half = lane >> 4;
  int col0 = lane & 15;
#pragma unroll
  for (int t = 0; t < 8; ++t) {
    int col = n0 + t * 16 + col0;
    float bv = bias[col];
#pragma unroll
    for (int r = 0; r < 8; ++r) {
      int row = m0 + wave * 16 + r + 8 * half;
      float v = acc[t][r] + bv;
      if (Cf) Cf[(size_t)row * N + col] = v;
      else    Cb[(size_t)row * N + col] = (bf16_t)v;
    }
  }
}

// ---------------------------------------------------------------------------
// RoPE (optional) + permute [B, S, Hn, DK] -> [B, Hn, S, DK].
__global__ void rope_permute(const bf16_t* __restrict__ lin,
                             const float* __restrict__ fcos,
                             const float* __restrict__ fsin,
                             bf16_t* __restrict__ out, int Hn, int do_rope) {
  size_t idx = (size_t)blockIdx.x * blockDim.x + threadIdx.x;
  size_t total = (size_t)B_ * S_ * Hn * (DK_ / 2);
  if (idx >= total) return;
  int i = (int)(idx % (DK_ / 2));
  size_t t = idx / (DK_ / 2);
  int h = (int)(t % Hn); t /= Hn;
  int s = (int)(t % S_);
  int b = (int)(t / S_);
  size_t src = (((size_t)b * S_ + s) * Hn + h) * DK_ + 2 * i;
  float xr = (float)lin[src];
  float xi = (float)lin[src + 1];
  float orr = xr, oi = xi;
  if (do_rope) {
    float c  = fcos[(size_t)s * (DK_ / 2) + i];
    float sn = fsin[(size_t)s * (DK_ / 2) + i];
    orr = xr * c - xi * sn;
    oi  = xr * sn + xi * c;
  }
  size_t dst = (((size_t)b * Hn + h) * S_ + s) * DK_ + 2 * i;
  out[dst]     = (bf16_t)orr;
  out[dst + 1] = (bf16_t)oi;
}

// ---------------------------------------------------------------------------
// Flash attention: grid (S_/128, B_*H_), block 256 (8 waves, one head/block).
__global__ void __launch_bounds__(256)
flash_attn(const bf16_t* __restrict__ Qt, const bf16_t* __restrict__ Kt,
           const bf16_t* __restrict__ Vt, bf16_t* __restrict__ Ob) {
  __shared__ bf16_t Ks[2][32][PAD128];   // [key][d], padded
  __shared__ bf16_t Vts[2][128][PAD32];  // transposed: [d][key], padded
  __shared__ bf16_t plds[8][16][PAD32];  // per-wave P staging, padded

  int tid  = threadIdx.x;
  int lane = tid & 31;
  int wave = tid >> 5;
  int bh = blockIdx.y;
  int b  = bh / H_;
  int h  = bh % H_;
  int hk = h / REP_;
  int q0 = blockIdx.x * 128 + wave * 16;

  const bf16_t* Qbase = Qt + ((size_t)(b * H_ + h) * S_ + q0) * DK_;
  const bf16_t* Kbase = Kt + (size_t)(b * HK_ + hk) * S_ * DK_;
  const bf16_t* Vbase = Vt + (size_t)(b * HK_ + hk) * S_ * DK_;

  v16bf qf[4];
#pragma unroll
  for (int j = 0; j < 4; ++j) qf[j] = frag_a_ld(Qbase + 32 * j, DK_, lane);

  v8f acc[8];
#pragma unroll
  for (int t = 0; t < 8; ++t) acc[t] = vzero8();
  float m_run[8], l_run[8];
#pragma unroll
  for (int r = 0; r < 8; ++r) { m_run[r] = -__builtin_inff(); l_run[r] = 0.0f; }

  const float scale = 0.08838834764831845f;  // 1/sqrt(DK)
  int half = lane >> 4;
  int col  = lane & 15;
  int kend_w = q0 + 16;                       // per-wave causal bound
  int nks = (blockIdx.x * 128 + 128) / 32;    // uniform over block

  auto stageK = [&](int buf, int k0) {        // 32x128: 512 chunks, 2/thread
#pragma unroll
    for (int j = 0; j < 2; ++j) {
      int idx = tid + j * 256;
      int key = idx >> 4;
      int ch  = idx & 15;
      async_copy_b128(Kbase + (size_t)(k0 + key) * DK_ + ch * 8, &Ks[buf][key][ch * 8]);
    }
  };
  auto stageV = [&](int buf, int k0) {        // transpose into [d][key]
#pragma unroll
    for (int j = 0; j < 2; ++j) {
      int idx = tid + j * 256;
      int key = idx & 31;
      int dch = idx >> 5;                     // 0..15, 8 d per chunk
      v8bf v = *(const v8bf*)(Vbase + (size_t)(k0 + key) * DK_ + dch * 8);
#pragma unroll
      for (int e = 0; e < 8; ++e) Vts[buf][dch * 8 + e][key] = v[e];
    }
  };

  stageK(0, 0);
  stageV(0, 0);

  for (int ks = 0; ks < nks; ++ks) {
    int k0  = ks * 32;
    int cur = ks & 1;
    ASYNC_WAIT0();
    __syncthreads();
    if (ks + 1 < nks) { stageK(cur ^ 1, k0 + 32); stageV(cur ^ 1, k0 + 32); }

    if (k0 < kend_w) {   // wave-uniform branch; EXEC stays all-ones inside
      // ---- scores: batch 8 K-fragments, then 8 WMMAs back-to-back
      v16bf kb[8];
#pragma unroll
      for (int j = 0; j < 4; ++j) {
        kb[2 * j]     = frag_b_ld(&Ks[cur][0][32 * j],  PAD128, lane);
        kb[2 * j + 1] = frag_b_ld(&Ks[cur][16][32 * j], PAD128, lane);
      }
      v8f s0 = vzero8(), s1 = vzero8();
#pragma unroll
      for (int j = 0; j < 4; ++j) {
        s0 = wmma_bf16(qf[j], kb[2 * j],     s0);
        s1 = wmma_bf16(qf[j], kb[2 * j + 1], s1);
      }
      // ---- online softmax (C layout: row = r + 8*half, col = lane&15)
#pragma unroll
      for (int r = 0; r < 8; ++r) {
        int qrow = q0 + r + 8 * half;
        float x0 = (k0 + col      <= qrow) ? s0[r] * scale : -__builtin_inff();
        float x1 = (k0 + 16 + col <= qrow) ? s1[r] * scale : -__builtin_inff();
        float tmax = fmaxf(x0, x1);
#pragma unroll
        for (int m = 1; m < 16; m <<= 1) tmax = fmaxf(tmax, __shfl_xor(tmax, m, 32));
        float m_new = fmaxf(m_run[r], tmax);
        float corr  = __expf(m_run[r] - m_new);
        float p0 = __expf(x0 - m_new);
        float p1 = __expf(x1 - m_new);
        float rs = p0 + p1;
#pragma unroll
        for (int m = 1; m < 16; m <<= 1) rs += __shfl_xor(rs, m, 32);
        l_run[r] = l_run[r] * corr + rs;
        m_run[r] = m_new;
#pragma unroll
        for (int t = 0; t < 8; ++t) acc[t][r] *= corr;
        int mrow = r + 8 * half;
        plds[wave][mrow][col]      = (bf16_t)p0;
        plds[wave][mrow][col + 16] = (bf16_t)p1;
      }
      DS_WAIT0();   // wave-private region: order DS store -> load within wave
      v16bf pf = frag_a_ld(&plds[wave][0][0], PAD32, lane);
      // ---- out += P(16x32) . V(32xDK): batch V-fragments then 8 WMMAs
      v16bf vf[8];
#pragma unroll
      for (int t = 0; t < 8; ++t) vf[t] = frag_b_ld(&Vts[cur][t * 16][0], PAD32, lane);
#pragma unroll
      for (int t = 0; t < 8; ++t) acc[t] = wmma_bf16(pf, vf[t], acc[t]);
    }
  }

  // ---- epilogue: normalize, store as [B, S, H*DK] bf16
#pragma unroll
  for (int t = 0; t < 8; ++t) {
#pragma unroll
    for (int r = 0; r < 8; ++r) {
      int row = q0 + r + 8 * half;
      int d   = t * 16 + (lane & 15);
      float o = acc[t][r] / l_run[r];
      Ob[((size_t)b * S_ + row) * D_ + h * DK_ + d] = (bf16_t)o;
    }
  }
}

// ---------------------------------------------------------------------------
extern "C" void kernel_launch(void* const* d_in, const int* in_sizes, int n_in,
                              void* d_out, int out_size, void* d_ws, size_t ws_size,
                              hipStream_t stream) {
  (void)in_sizes; (void)n_in; (void)out_size; (void)ws_size;
  const float* x    = (const float*)d_in[0];
  const float* fcos = (const float*)d_in[1];
  const float* fsin = (const float*)d_in[2];
  const float* wq_w = (const float*)d_in[3];
  const float* wq_b = (const float*)d_in[4];
  const float* wk_w = (const float*)d_in[5];
  const float* wk_b = (const float*)d_in[6];
  const float* wv_w = (const float*)d_in[7];
  const float* wv_b = (const float*)d_in[8];
  const float* wo_w = (const float*)d_in[9];
  const float* wo_b = (const float*)d_in[10];

  const size_t NX  = (size_t)B_ * S_ * D_;
  const size_t NWQ = (size_t)D_ * D_;
  const size_t NWK = (size_t)KD_ * D_;
  const size_t NKL = (size_t)B_ * S_ * KD_;

  char* w = (char*)d_ws;
  auto alloc = [&](size_t elems) {
    bf16_t* p = (bf16_t*)w;
    w += (elems * sizeof(bf16_t) + 255) & ~(size_t)255;
    return p;
  };
  bf16_t* xbf  = alloc(NX);
  bf16_t* wqbf = alloc(NWQ);
  bf16_t* wkbf = alloc(NWK);
  bf16_t* wvbf = alloc(NWK);
  bf16_t* wobf = alloc(NWQ);
  bf16_t* qlin = alloc(NX);
  bf16_t* klin = alloc(NKL);
  bf16_t* vlin = alloc(NKL);
  bf16_t* qt   = alloc(NX);
  bf16_t* kt   = alloc(NKL);
  bf16_t* vt   = alloc(NKL);
  bf16_t* obf  = alloc(NX);

  cvt_f32_bf16<<<dim3((unsigned)((NX  + 255) / 256)), 256, 0, stream>>>(x,    xbf,  NX);
  cvt_f32_bf16<<<dim3((unsigned)((NWQ + 255) / 256)), 256, 0, stream>>>(wq_w, wqbf, NWQ);
  cvt_f32_bf16<<<dim3((unsigned)((NWK + 255) / 256)), 256, 0, stream>>>(wk_w, wkbf, NWK);
  cvt_f32_bf16<<<dim3((unsigned)((NWK + 255) / 256)), 256, 0, stream>>>(wv_w, wvbf, NWK);
  cvt_f32_bf16<<<dim3((unsigned)((NWQ + 255) / 256)), 256, 0, stream>>>(wo_w, wobf, NWQ);

  const int M = B_ * S_;
  dim3 gq(D_ / 128,  M / 128);
  dim3 gk(KD_ / 128, M / 128);
  gemm_bf16_nt<<<gq, 256, 0, stream>>>(xbf, wqbf, wq_b, nullptr, qlin, M, D_,  D_);
  gemm_bf16_nt<<<gk, 256, 0, stream>>>(xbf, wkbf, wk_b, nullptr, klin, M, KD_, D_);
  gemm_bf16_nt<<<gk, 256, 0, stream>>>(xbf, wvbf, wv_b, nullptr, vlin, M, KD_, D_);

  size_t nqp = (size_t)B_ * S_ * H_  * (DK_ / 2);
  size_t nkp = (size_t)B_ * S_ * HK_ * (DK_ / 2);
  rope_permute<<<dim3((unsigned)((nqp + 255) / 256)), 256, 0, stream>>>(qlin, fcos, fsin, qt, H_,  1);
  rope_permute<<<dim3((unsigned)((nkp + 255) / 256)), 256, 0, stream>>>(klin, fcos, fsin, kt, HK_, 1);
  rope_permute<<<dim3((unsigned)((nkp + 255) / 256)), 256, 0, stream>>>(vlin, fcos, fsin, vt, HK_, 0);

  flash_attn<<<dim3(S_ / 128, B_ * H_), 256, 0, stream>>>(qt, kt, vt, obf);

  gemm_bf16_nt<<<gq, 256, 0, stream>>>(obf, wobf, wo_b, (float*)d_out, nullptr, M, D_, D_);
}
